// LlamaSemiARHead_88545045774991
// MI455X (gfx1250) — matrix operand order repaired and checked by reading, not compiled, not run on hardware
//
#include <hip/hip_runtime.h>

// ---------------------------------------------------------------------------
// CDNA5 / gfx1250. wave32. Matmuls via v_wmma_f32_16x16x32_bf16 (f32 acc).
// GEMM tiles staged into LDS by the Tensor Data Mover (double-buffered,
// 64-deep K steps), fragments preloaded to keep the matrix pipe saturated.
// ---------------------------------------------------------------------------

typedef __attribute__((ext_vector_type(16))) __bf16 v16bf;
typedef __attribute__((ext_vector_type(8)))  float  v8f;

union FragB16 {
    v16bf v;
    uint4 u[2];
    unsigned short s[16];
};

__device__ __forceinline__ v8f wmma_bf16(v16bf a, v16bf b, v8f c) {
    return __builtin_amdgcn_wmma_f32_16x16x32_bf16(false, a, false, b, (short)0, c,
                                                   false, false);
}

__device__ __forceinline__ unsigned short f2bf(float f) {
    unsigned int u = __float_as_uint(f);
    unsigned int rnd = 0x7FFFu + ((u >> 16) & 1u);   // round-to-nearest-even
    return (unsigned short)((u + rnd) >> 16);
}

#ifndef __has_builtin
#define __has_builtin(x) 0
#endif
#if defined(__HIP_DEVICE_COMPILE__) && \
    __has_builtin(__builtin_amdgcn_tensor_load_to_lds) && \
    __has_builtin(__builtin_amdgcn_s_wait_tensorcnt)
#define USE_TDM 1
#else
#define USE_TDM 0
#endif

#if USE_TDM
typedef __attribute__((ext_vector_type(4))) unsigned int u32x4;
typedef __attribute__((ext_vector_type(4))) int i32x4;
typedef __attribute__((ext_vector_type(8))) int i32x8;

// DMA a 2D tile (tile_w x tile_h elements of 2 bytes, row stride in elements)
// from global memory into LDS via the Tensor Data Mover (TENSORcnt).
__device__ __forceinline__ void tdm_load_tile_2d(const unsigned short* gsrc,
                                                 unsigned short* lds_dst,
                                                 unsigned tile_w, unsigned tile_h,
                                                 unsigned long long row_stride) {
    unsigned long long ga = (unsigned long long)(const void*)gsrc;
    unsigned lds_addr = (unsigned)(unsigned long long)(void*)lds_dst;
    u32x4 g0;
    g0[0] = 1u;                                               // count=1, user desc
    g0[1] = lds_addr;                                         // lds_addr
    g0[2] = (unsigned)ga;                                     // global_addr lo
    g0[3] = (unsigned)((ga >> 32) & 0x01FFFFFFu) | 0x80000000u; // addr hi | type=2
    i32x8 g1;
    g1[0] = 0x00010000;                       // workgroup_mask=0, data_size=2B
    g1[1] = (int)(tile_w << 16);              // tensor_dim0 lo16 (== tile_w)
    g1[2] = (int)(tile_h << 16);              // tensor_dim0 hi=0 | tensor_dim1 lo16
    g1[3] = (int)(tile_w << 16);              // tensor_dim1 hi=0 | tile_dim0
    g1[4] = (int)tile_h;                      // tile_dim1 | tile_dim2=0
    g1[5] = (int)(unsigned)row_stride;        // tensor_dim0_stride lo32
    g1[6] = (int)(unsigned)(row_stride >> 32);// stride hi16 | dim1_stride lo16 (0)
    g1[7] = 0;
    i32x4 z4 = {0, 0, 0, 0};
    i32x8 z8 = {0, 0, 0, 0, 0, 0, 0, 0};
    __builtin_amdgcn_tensor_load_to_lds(g0, g1, z4, z4, z8, 0);
}
#endif

// ---------------------------------------------------------------------------
// fp32 -> bf16 cast
// ---------------------------------------------------------------------------
__global__ void cast_f32_bf16(const float* __restrict__ in,
                              unsigned short* __restrict__ out, int n) {
    int i = blockIdx.x * blockDim.x + threadIdx.x;
    if (i < n) out[i] = f2bf(in[i]);
}

// ---------------------------------------------------------------------------
// GEMM: out[M,N] = A[M,Kd](bf16) @ W[N,Kd]^T(bf16) + bias, optional SiLU.
// 256 threads = 8 waves (4x2), block tile 256x128, wave tile 64x64.
// K staged 64-deep: 32 wmma per wave between barriers, frags preloaded.
// ---------------------------------------------------------------------------
#define GBM 256
#define GBN 128
#define GBK 64
#define ASZ (GBM * GBK)
#define BSZ (GBN * GBK)

__global__ __launch_bounds__(256) void gemm_bf16_wmma(
    const unsigned short* __restrict__ A, const unsigned short* __restrict__ W,
    const float* __restrict__ bias, float* __restrict__ outF,
    unsigned short* __restrict__ outB, int M, int N, int Kd, int act) {
#if USE_TDM
    __shared__ unsigned short lds_a[2 * ASZ];   // 64 KB (double buffered)
    __shared__ unsigned short lds_b[2 * BSZ];   // 32 KB
#else
    __shared__ unsigned short lds_a[ASZ];
    __shared__ unsigned short lds_b[BSZ];
#endif

    const int tid  = threadIdx.x;
    const int lane = tid & 31;
    const int wave = tid >> 5;
    const int r    = lane & 15;
    const int half = lane >> 4;
    const int wm   = wave >> 1;          // 0..3
    const int wn   = wave & 1;           // 0..1
    const size_t m0 = (size_t)blockIdx.y * GBM;
    const size_t n0 = (size_t)blockIdx.x * GBN;

    v8f acc[4][4] = {};

#if USE_TDM
    const int T = Kd / GBK;
    if (wave == 0) {
        tdm_load_tile_2d(A + m0 * Kd, lds_a, GBK, GBM, (unsigned long long)Kd);
        tdm_load_tile_2d(W + n0 * Kd, lds_b, GBK, GBN, (unsigned long long)Kd);
    }
    for (int s = 0; s < T; s++) {
        const int cur = s & 1;
        if (wave == 0) {
            if (s + 1 < T) {
                int knx = (s + 1) * GBK;
                tdm_load_tile_2d(A + m0 * Kd + knx, lds_a + (1 - cur) * ASZ,
                                 GBK, GBM, (unsigned long long)Kd);
                tdm_load_tile_2d(W + n0 * Kd + knx, lds_b + (1 - cur) * BSZ,
                                 GBK, GBN, (unsigned long long)Kd);
                __builtin_amdgcn_s_wait_tensorcnt(2);   // pair(s) complete
            } else {
                __builtin_amdgcn_s_wait_tensorcnt(0);
            }
        }
        __syncthreads();
        const unsigned short* la = lds_a + cur * ASZ;
        const unsigned short* lb = lds_b + cur * BSZ;
#else
    for (int k0 = 0; k0 < Kd; k0 += GBK) {
        // stage A tile 256x64 (one row of 64 bf16 per thread)
        {
            const uint4* g = reinterpret_cast<const uint4*>(A + (m0 + tid) * Kd + k0);
            uint4* s4 = reinterpret_cast<uint4*>(lds_a + tid * GBK);
#pragma unroll
            for (int q = 0; q < 8; q++) s4[q] = g[q];
        }
        // stage B tile 128x64 (half row of 32 bf16 per thread)
        {
            int row = tid >> 1, h2 = tid & 1;
            const uint4* g =
                reinterpret_cast<const uint4*>(W + (n0 + row) * Kd + k0 + h2 * 32);
            uint4* s4 = reinterpret_cast<uint4*>(lds_b + row * GBK + h2 * 32);
#pragma unroll
            for (int q = 0; q < 4; q++) s4[q] = g[q];
        }
        __syncthreads();
        const unsigned short* la = lds_a;
        const unsigned short* lb = lds_b;
#endif
#pragma unroll
        for (int kk = 0; kk < GBK; kk += 32) {
            FragB16 af[4], bfr[4];
#pragma unroll
            for (int mi = 0; mi < 4; mi++) {
                const unsigned short* p =
                    la + (wm * 64 + mi * 16 + r) * GBK + kk + half * 8;
                af[mi].u[0] = *reinterpret_cast<const uint4*>(p);
                af[mi].u[1] = *reinterpret_cast<const uint4*>(p + 16);
            }
#pragma unroll
            for (int nj = 0; nj < 4; nj++) {
                const unsigned short* p =
                    lb + (wn * 64 + nj * 16 + r) * GBK + kk + half * 8;
                bfr[nj].u[0] = *reinterpret_cast<const uint4*>(p);
                bfr[nj].u[1] = *reinterpret_cast<const uint4*>(p + 16);
            }
#pragma unroll
            for (int nj = 0; nj < 4; nj++)
#pragma unroll
                for (int mi = 0; mi < 4; mi++)
                    acc[mi][nj] = wmma_bf16(af[mi].v, bfr[nj].v, acc[mi][nj]);
        }
        __syncthreads();
    }

    // epilogue: element v of lane (r,half) is row half*8+v, col r of each frag
    for (int mi = 0; mi < 4; mi++) {
        for (int nj = 0; nj < 4; nj++) {
            for (int v = 0; v < 8; v++) {
                size_t row = m0 + wm * 64 + mi * 16 + half * 8 + v;
                size_t col = n0 + wn * 64 + nj * 16 + r;
                float val = acc[mi][nj][v] + bias[col];
                if (act) val = val / (1.0f + __expf(-val));   // SiLU
                if (outF) outF[row * N + col] = val;
                if (outB) outB[row * N + col] = f2bf(val);
            }
        }
    }
}

// ---------------------------------------------------------------------------
// RoPE + cast to bf16.  Layout (S, nheads*128); pairs (j, j+64) share angle.
// ---------------------------------------------------------------------------
__global__ void rope_cast(const float* __restrict__ in,
                          unsigned short* __restrict__ out, int S, int nheads) {
    int i = blockIdx.x * blockDim.x + threadIdx.x;
    int total = S * nheads * 64;
    if (i >= total) return;
    int j = i & 63;
    int h = (i >> 6) % nheads;
    int s = i / (64 * nheads);
    const float LN1E4 = 9.210340371976184f;   // ln(10000)
    float theta = (float)s * __expf(-(float)(2 * j) * (LN1E4 / 128.0f));
    float sn, cs;
    __sincosf(theta, &sn, &cs);
    size_t base = (size_t)s * (nheads * 128) + (size_t)h * 128;
    float x1 = in[base + j];
    float x2 = in[base + j + 64];
    out[base + j]      = f2bf(x1 * cs - x2 * sn);
    out[base + j + 64] = f2bf(x2 * cs + x1 * sn);
}

// ---------------------------------------------------------------------------
// Flash attention. 4 waves/block; wave w owns 16 query rows (i0b + w*16),
// all share one (h,g) pair and cooperatively stage the V^T tile in LDS.
// Qb: (S,1024) bf16 rope'd; Kb/Vb: (S,512) bf16. Opart: (H*G,S,128) f32.
// ---------------------------------------------------------------------------
__global__ __launch_bounds__(128) void attn_flash(
    const unsigned short* __restrict__ Qb, const unsigned short* __restrict__ Kb,
    const unsigned short* __restrict__ Vb, float* __restrict__ Opart, int S) {
    __shared__ unsigned short lds_vt[128 * 32];        // V^T tile [d][j], 8 KB
    __shared__ unsigned short lds_p[4 * 16 * 32];      // P tile per wave, 4 KB

    const int tid  = threadIdx.x;
    const int lane = tid & 31;
    const int wave = tid >> 5;
    const int r    = lane & 15;
    const int half = lane >> 4;
    const int i0b  = blockIdx.x * 64;
    const int i0   = i0b + wave * 16;
    const int h    = blockIdx.y;
    const int g    = blockIdx.z;
    const int HD = 1024, GD = 512;
    unsigned short* ldsp = lds_p + wave * (16 * 32);

    // Q fragments: 4 K-chunks of 32 over D=128 (A layout)
    FragB16 qf[4];
    for (int dc = 0; dc < 4; dc++) {
        const unsigned short* p =
            Qb + (size_t)(i0 + r) * HD + h * 128 + dc * 32 + half * 8;
        qf[dc].u[0] = *reinterpret_cast<const uint4*>(p);
        qf[dc].u[1] = *reinterpret_cast<const uint4*>(p + 16);
    }

    v8f o[8] = {};
    float mrow[8], lrow[8];
    for (int v = 0; v < 8; v++) { mrow[v] = -1e30f; lrow[v] = 0.0f; }
    const float scl = 0.08838834764831845f;   // 1/sqrt(128)
    const int jmaxw = i0 + 15;
    const int jmaxb = i0b + 63;

    for (int j0 = 0; j0 <= jmaxb; j0 += 32) {
        // cooperative V^T staging: thread -> (row j, 32-wide d segment)
        {
            int jj = tid & 31, dseg = tid >> 5;
            const unsigned short* vr =
                Vb + (size_t)(j0 + jj) * GD + g * 128 + dseg * 32;
            for (int d = 0; d < 32; d++)
                lds_vt[(dseg * 32 + d) * 32 + jj] = vr[d];
        }
        // prefetch next step's K/V rows into cache (global_prefetch_b8)
        if (j0 + 32 <= jmaxb) {
            __builtin_prefetch(Kb + (size_t)(j0 + 32 + lane) * GD + g * 128, 0, 3);
            __builtin_prefetch(Vb + (size_t)(j0 + 32 + lane) * GD + g * 128, 0, 3);
        }

        const bool active = (j0 <= jmaxw);   // causal tail idle per wave
        float scale[8];
        if (active) {
            // scores: S = Q K^T for two 16-wide j sub-tiles
            v8f sfr[2];
            for (int jt = 0; jt < 2; jt++) {
                v8f s = {};
                for (int dc = 0; dc < 4; dc++) {
                    FragB16 kf;   // B layout: lane r = column j, K runs over d
                    const unsigned short* p = Kb + (size_t)(j0 + jt * 16 + r) * GD +
                                              g * 128 + dc * 32 + half * 8;
                    kf.u[0] = *reinterpret_cast<const uint4*>(p);
                    kf.u[1] = *reinterpret_cast<const uint4*>(p + 16);
                    s = wmma_bf16(qf[dc].v, kf.v, s);
                }
                sfr[jt] = s;
            }
            // causal mask + scale + row max
            float rowmax[8];
            for (int v = 0; v < 8; v++) rowmax[v] = -1e30f;
            for (int jt = 0; jt < 2; jt++) {
                int jcol = j0 + jt * 16 + r;
                for (int v = 0; v < 8; v++) {
                    int irow = i0 + half * 8 + v;
                    float sv = sfr[jt][v] * scl;
                    if (jcol > irow) sv = -1e30f;
                    sfr[jt][v] = sv;
                    rowmax[v]  = fmaxf(rowmax[v], sv);
                }
            }
            for (int off = 8; off >= 1; off >>= 1)
                for (int v = 0; v < 8; v++)
                    rowmax[v] = fmaxf(rowmax[v], __shfl_xor(rowmax[v], off, 32));

            float rs[8];
            for (int v = 0; v < 8; v++) {
                float mnew = fmaxf(mrow[v], rowmax[v]);
                scale[v]   = __expf(mrow[v] - mnew);
                mrow[v]    = mnew;
                rs[v]      = 0.0f;
            }
            // P = exp(S - m): C-layout -> A-layout bridge via LDS
            for (int jt = 0; jt < 2; jt++) {
                for (int v = 0; v < 8; v++) {
                    float pv = __expf(sfr[jt][v] - mrow[v]);
                    rs[v] += pv;
                    ldsp[(half * 8 + v) * 32 + jt * 16 + r] = f2bf(pv);
                }
            }
            for (int off = 8; off >= 1; off >>= 1)
                for (int v = 0; v < 8; v++) rs[v] += __shfl_xor(rs[v], off, 32);
            for (int v = 0; v < 8; v++) lrow[v] = lrow[v] * scale[v] + rs[v];
        }
        __syncthreads();   // V^T + P visible

        if (active) {
            for (int n = 0; n < 8; n++)
                for (int v = 0; v < 8; v++) o[n][v] *= scale[v];
            // preload P fragment and all 8 V fragments, then burst 8 wmma
            FragB16 pf, vfr[8];
            {
                const unsigned short* p = ldsp + r * 32 + half * 8;
                pf.u[0] = *reinterpret_cast<const uint4*>(p);
                pf.u[1] = *reinterpret_cast<const uint4*>(p + 16);
            }
#pragma unroll
            for (int n = 0; n < 8; n++) {
                const unsigned short* p = lds_vt + (n * 16 + r) * 32 + half * 8;
                vfr[n].u[0] = *reinterpret_cast<const uint4*>(p);
                vfr[n].u[1] = *reinterpret_cast<const uint4*>(p + 16);
            }
#pragma unroll
            for (int n = 0; n < 8; n++)
                o[n] = wmma_bf16(pf.v, vfr[n].v, o[n]);
        }
        __syncthreads();   // protect tiles before restage
    }

    // normalize and store per-(h,g) partial output
    size_t obase = ((size_t)(h * 4 + g) * S + i0) * 128;
    for (int v = 0; v < 8; v++) {
        float inv = 1.0f / lrow[v];
        for (int n = 0; n < 8; n++)
            Opart[obase + (size_t)(half * 8 + v) * 128 + n * 16 + r] = o[n][v] * inv;
    }
}

// ---------------------------------------------------------------------------
// Sum partial attention outputs over the 4 kv-groups -> attn bf16 (S, 1024)
// ---------------------------------------------------------------------------
__global__ void reduce_groups(const float* __restrict__ Opart,
                              unsigned short* __restrict__ attnB, int S) {
    int i = blockIdx.x * blockDim.x + threadIdx.x;   // over S*1024
    if (i >= S * 1024) return;
    int d = i & 127;
    int h = (i >> 7) & 7;
    int s = i >> 10;
    float acc = 0.0f;
    for (int g = 0; g < 4; g++)
        acc += Opart[(((size_t)(h * 4 + g) * S + s) << 7) + d];
    attnB[i] = f2bf(acc);
}

// ---------------------------------------------------------------------------
// Host-side orchestration
// ---------------------------------------------------------------------------
extern "C" void kernel_launch(void* const* d_in, const int* in_sizes, int n_in,
                              void* d_out, int out_size, void* d_ws, size_t ws_size,
                              hipStream_t stream) {
    const int S = 2048, DIM = 1024, H = 8, G = 4, D = 128;
    const int HD = H * D;        // 1024
    const int GD = G * D;        // 512
    const int FF = DIM * 4;      // 4096
    const int NO = 4 * DIM;      // 4096

    const float* x  = (const float*)d_in[0];
    const float* Wq = (const float*)d_in[1];
    const float* bq = (const float*)d_in[2];
    const float* Wk = (const float*)d_in[3];
    const float* bk = (const float*)d_in[4];
    const float* Wv = (const float*)d_in[5];
    const float* bv = (const float*)d_in[6];
    const float* Wo = (const float*)d_in[7];
    const float* bo = (const float*)d_in[8];
    const float* W1 = (const float*)d_in[9];
    const float* b1 = (const float*)d_in[10];
    const float* W2 = (const float*)d_in[11];
    const float* b2 = (const float*)d_in[12];
    float* outp = (float*)d_out;

    char* wp = (char*)d_ws;
    auto carve = [&](size_t bytes) -> char* {
        char* p = wp;
        wp += (bytes + 255) & ~(size_t)255;
        return p;
    };
    unsigned short* xb   = (unsigned short*)carve((size_t)S * DIM * 2);
    unsigned short* wqb  = (unsigned short*)carve((size_t)HD * DIM * 2);
    unsigned short* wkb  = (unsigned short*)carve((size_t)GD * DIM * 2);
    unsigned short* wvb  = (unsigned short*)carve((size_t)GD * DIM * 2);
    unsigned short* wob  = (unsigned short*)carve((size_t)DIM * HD * 2);
    unsigned short* w1b  = (unsigned short*)carve((size_t)FF * DIM * 2);
    unsigned short* w2b  = (unsigned short*)carve((size_t)NO * FF * 2);
    float*          qf   = (float*)carve((size_t)S * HD * 4);
    float*          kf   = (float*)carve((size_t)S * GD * 4);
    unsigned short* qb   = (unsigned short*)carve((size_t)S * HD * 2);
    unsigned short* kb   = (unsigned short*)carve((size_t)S * GD * 2);
    unsigned short* vb   = (unsigned short*)carve((size_t)S * GD * 2);
    float*          op   = (float*)carve((size_t)H * G * S * D * 4);
    unsigned short* attb = (unsigned short*)carve((size_t)S * HD * 2);
    unsigned short* ob   = (unsigned short*)carve((size_t)S * DIM * 2);
    unsigned short* h1b  = (unsigned short*)carve((size_t)S * FF * 2);
    (void)ws_size; (void)n_in; (void)in_sizes; (void)out_size;

    auto cast = [&](const float* src, unsigned short* dst, int n) {
        cast_f32_bf16<<<(n + 255) / 256, 256, 0, stream>>>(src, dst, n);
    };
    cast(x,  xb,  S * DIM);
    cast(Wq, wqb, HD * DIM);
    cast(Wk, wkb, GD * DIM);
    cast(Wv, wvb, GD * DIM);
    cast(Wo, wob, DIM * HD);
    cast(W1, w1b, FF * DIM);
    cast(W2, w2b, NO * FF);

    auto gemm = [&](const unsigned short* A, const unsigned short* W,
                    const float* bias, float* oF, unsigned short* oB,
                    int M, int N, int Kd, int act) {
        dim3 grid(N / GBN, M / GBM);
        gemm_bf16_wmma<<<grid, 256, 0, stream>>>(A, W, bias, oF, oB, M, N, Kd, act);
    };

    // projections
    gemm(xb, wqb, bq, qf, nullptr, S, HD, DIM, 0);
    gemm(xb, wkb, bk, kf, nullptr, S, GD, DIM, 0);
    gemm(xb, wvb, bv, nullptr, vb, S, GD, DIM, 0);

    // rotary + bf16 cast
    {
        int nq = S * H * 64;
        rope_cast<<<(nq + 255) / 256, 256, 0, stream>>>(qf, qb, S, H);
        int nk = S * G * 64;
        rope_cast<<<(nk + 255) / 256, 256, 0, stream>>>(kf, kb, S, G);
    }

    // attention: grid (64-row i-blocks, heads, groups), 4 waves per block
    {
        dim3 grid(S / 64, H, G);
        attn_flash<<<grid, 128, 0, stream>>>(qb, kb, vb, op, S);
        int n = S * HD;
        reduce_groups<<<(n + 255) / 256, 256, 0, stream>>>(op, attb, S);
    }

    // output projection + MLP
    gemm(attb, wob, bo, nullptr, ob, S, DIM, HD, 0);
    gemm(ob,  w1b, b1, nullptr, h1b, S, FF, DIM, 1);   // SiLU
    gemm(h1b, w2b, b2, outp, nullptr, S, NO, FF, 0);
}
